// Attention_61332132987140
// MI455X (gfx1250) — matrix operand order repaired and checked by reading, not compiled, not run on hardware
//
#include <hip/hip_runtime.h>
#include <hip/hip_bf16.h>
#include <math.h>

#define DIMX 1024
#define HEADS 16
#define DH 64
#define B_SZ 2
#define N_SEQ 2048
#define ROWS (B_SZ * N_SEQ)      // 4096
#define QKV_COLS (3 * DIMX)      // 3072

typedef __attribute__((ext_vector_type(16))) _Float16 v16h;
typedef __attribute__((ext_vector_type(8)))  float    v8f;
typedef __attribute__((ext_vector_type(4)))  int          v4i;
typedef __attribute__((ext_vector_type(4)))  unsigned int u32x4;
typedef __attribute__((ext_vector_type(8)))  int          i32x8;
typedef __attribute__((ext_vector_type(4)))  int          i32x4;

#ifndef __has_builtin
#define __has_builtin(x) 0
#endif
#if __has_builtin(__builtin_amdgcn_global_load_async_to_lds_b128)
#define HAVE_ASYNC 1
#else
#define HAVE_ASYNC 0
#endif
#if __has_builtin(__builtin_amdgcn_s_wait_asynccnt)
#define HAVE_ASYNC_WAIT 1
#else
#define HAVE_ASYNC_WAIT 0
#endif
#if __has_builtin(__builtin_amdgcn_tensor_load_to_lds) && __has_builtin(__builtin_amdgcn_s_wait_tensorcnt)
#define HAVE_TDM 1
#else
#define HAVE_TDM 0
#endif
#if __has_builtin(__builtin_amdgcn_update_dpp)
#define HAVE_DPP 1
#else
#define HAVE_DPP 0
#endif

// ---------------------------------------------------------------------------
// async copy helpers (ASYNCcnt-tracked global->LDS; sync fallback)
// builtin signature: (as1 v4i* gsrc, as3 v4i* ldst, imm offset, imm cpol)
// ---------------------------------------------------------------------------
typedef __attribute__((address_space(1))) v4i as1_v4i;
typedef __attribute__((address_space(3))) v4i as3_v4i;

static __device__ inline void async_copy16(void* lds, const void* g) {
#if HAVE_ASYNC
  __builtin_amdgcn_global_load_async_to_lds_b128(
      (as1_v4i*)g, (as3_v4i*)lds, 0, 0);
#else
  *(uint4*)lds = *(const uint4*)g;
#endif
}

static __device__ inline void async_wait(int n) {
#if HAVE_ASYNC
#if HAVE_ASYNC_WAIT
  if (n == 0) __builtin_amdgcn_s_wait_asynccnt(0);
  else if (n == 1) __builtin_amdgcn_s_wait_asynccnt(1);
  else __builtin_amdgcn_s_wait_asynccnt(2);
#else
  if (n == 0) asm volatile("s_wait_asynccnt 0x0" ::: "memory");
  else if (n == 1) asm volatile("s_wait_asynccnt 0x1" ::: "memory");
  else asm volatile("s_wait_asynccnt 0x2" ::: "memory");
#endif
#else
  (void)n;
#endif
}

static __device__ inline void tensor_wait(int n) {
#if HAVE_TDM
  if (n == 0) __builtin_amdgcn_s_wait_tensorcnt(0);
  else __builtin_amdgcn_s_wait_tensorcnt(1);
#else
  (void)n;
#endif
}

// ---------------------------------------------------------------------------
// WMMA helpers (CDNA5 gfx1250, wave32). D = A(16x32 f16) * B(32x16 f16) + C(f32)
// ---------------------------------------------------------------------------
static __device__ inline v8f wmma32(v16h a, v16h b, v8f c) {
  return __builtin_amdgcn_wmma_f32_16x16x32_f16(
      false, a, false, b, (short)0, c, false, false);
}

// A-matrix 16x32 f16 fragment (ISA layout):
// lanes 0-15  (M=lane):    halves 0..7 = K0..7,  halves 8..15 = K16..23
// lanes 16-31 (M=lane-16): halves 0..7 = K8..15, halves 8..15 = K24..31
// caller guarantees 16B alignment of every 8-half chunk (ld in {40,64,72})
static __device__ inline v16h load_frag_a(const _Float16* __restrict__ src, int ld) {
  const int lane = threadIdx.x & 31;
  const int m = lane & 15;
  const int kb = (lane < 16) ? 0 : 8;
  const _Float16* row = src + (size_t)m * ld;
  v16h a;
  ((uint4*)&a)[0] = *(const uint4*)(row + kb);
  ((uint4*)&a)[1] = *(const uint4*)(row + 16 + kb);
  return a;
}

// B-matrix 32x16 f16 fragment; source column-major (col n contiguous over K).
// lanes 0-15: col=lane, K0..7,K16..23; lanes 16-31: col=lane-16, K8..15,K24..31.
static __device__ inline v16h load_frag_b(const _Float16* __restrict__ w, int ld) {
  const int lane = threadIdx.x & 31;
  const int n = lane & 15;
  const int kb = (lane < 16) ? 0 : 8;
  const _Float16* col = w + (size_t)n * ld + kb;
  v16h b;
  ((uint4*)&b)[0] = *(const uint4*)(col);
  ((uint4*)&b)[1] = *(const uint4*)(col + 16);
  return b;
}

// C/D fragment store: lane 0-15 VGPR r -> (M=r, N=lane); lanes 16-31 -> M=r+8.
static __device__ inline void store_c_lds(float* dst, int ld, v8f c) {
  const int lane = threadIdx.x & 31;
  const int n = lane & 15;
  const int mb = (lane < 16) ? 0 : 8;
#pragma unroll
  for (int r = 0; r < 8; ++r) dst[(mb + r) * ld + n] = c[r];
}

// ---------------------------------------------------------------------------
// 16-lane row reductions: DPP (VALU-only) if available, else shuffles
// ---------------------------------------------------------------------------
#if HAVE_DPP
static __device__ inline float red16_max(float v) {
  v = fmaxf(v, __int_as_float(__builtin_amdgcn_update_dpp(0, __float_as_int(v), 0xB1, 0xF, 0xF, true)));
  v = fmaxf(v, __int_as_float(__builtin_amdgcn_update_dpp(0, __float_as_int(v), 0x4E, 0xF, 0xF, true)));
  v = fmaxf(v, __int_as_float(__builtin_amdgcn_update_dpp(0, __float_as_int(v), 0x141, 0xF, 0xF, true)));
  v = fmaxf(v, __int_as_float(__builtin_amdgcn_update_dpp(0, __float_as_int(v), 0x140, 0xF, 0xF, true)));
  return v;
}
static __device__ inline float red16_sum(float v) {
  v += __int_as_float(__builtin_amdgcn_update_dpp(0, __float_as_int(v), 0xB1, 0xF, 0xF, true));
  v += __int_as_float(__builtin_amdgcn_update_dpp(0, __float_as_int(v), 0x4E, 0xF, 0xF, true));
  v += __int_as_float(__builtin_amdgcn_update_dpp(0, __float_as_int(v), 0x141, 0xF, 0xF, true));
  v += __int_as_float(__builtin_amdgcn_update_dpp(0, __float_as_int(v), 0x140, 0xF, 0xF, true));
  return v;
}
#else
static __device__ inline float red16_max(float v) {
#pragma unroll
  for (int m = 1; m <= 8; m <<= 1) v = fmaxf(v, __shfl_xor(v, m, 32));
  return v;
}
static __device__ inline float red16_sum(float v) {
#pragma unroll
  for (int m = 1; m <= 8; m <<= 1) v += __shfl_xor(v, m, 32);
  return v;
}
#endif

static __device__ inline float wave_sum(float v) {
#pragma unroll
  for (int m = 16; m >= 1; m >>= 1) v += __shfl_xor(v, m, 32);
  return v;
}

// ---------------------------------------------------------------------------
// K0: f32 -> f16 conversion (weights)
// ---------------------------------------------------------------------------
__global__ __launch_bounds__(256)
void cvt_f16_kernel(const float* __restrict__ src, _Float16* __restrict__ dst, int n) {
  int i = blockIdx.x * 256 + threadIdx.x;
  if (i < n) dst[i] = (_Float16)src[i];
}

// ---------------------------------------------------------------------------
// K1: fused RMSNorm (F.normalize * sqrt(dim) * gamma) + per-head sigmoid gates
// ---------------------------------------------------------------------------
__global__ __launch_bounds__(256)
void rmsnorm_gates_kernel(const float* __restrict__ x, const float* __restrict__ gamma,
                          const float* __restrict__ wg, const float* __restrict__ bg,
                          _Float16* __restrict__ xnh, float* __restrict__ gates) {
  __shared__ float xs[DIMX];
  __shared__ float sred[8];
  const int row = blockIdx.x;
  const int tid = threadIdx.x;

  const float4 xv = ((const float4*)(x + (size_t)row * DIMX))[tid];
  float ss = xv.x * xv.x + xv.y * xv.y + xv.z * xv.z + xv.w * xv.w;
  ss = wave_sum(ss);
  if ((tid & 31) == 0) sred[tid >> 5] = ss;
  __syncthreads();
  float tot = 0.f;
#pragma unroll
  for (int i = 0; i < 8; ++i) tot += sred[i];
  const float rs = 32.0f / fmaxf(sqrtf(tot), 1e-12f);  // sqrt(1024)=32

  const float4 gv = ((const float4*)gamma)[tid];
  float4 xn;
  xn.x = xv.x * rs * gv.x;
  xn.y = xv.y * rs * gv.y;
  xn.z = xv.z * rs * gv.z;
  xn.w = xv.w * rs * gv.w;
  const int d = tid * 4;
  xs[d + 0] = xn.x; xs[d + 1] = xn.y; xs[d + 2] = xn.z; xs[d + 3] = xn.w;
  _Float16* dst = xnh + (size_t)row * DIMX + d;
  dst[0] = (_Float16)xn.x; dst[1] = (_Float16)xn.y;
  dst[2] = (_Float16)xn.z; dst[3] = (_Float16)xn.w;
  __syncthreads();

  const int wv = tid >> 5, ln = tid & 31;
#pragma unroll
  for (int hh = 0; hh < 2; ++hh) {
    const int h = wv * 2 + hh;
    float s = 0.f;
    for (int i = ln; i < DIMX; i += 32) s += xs[i] * wg[(size_t)h * DIMX + i];
    s = wave_sum(s);
    if (ln == 0) gates[(size_t)row * HEADS + h] = 1.f / (1.f + __expf(-(s + bg[h])));
  }
}

// ---------------------------------------------------------------------------
// K2/K4: 64x64-tile WMMA GEMM  C[m][j] = sum_k A[m][k] * W[j][k]
// Double-buffered LDS staging: A tile via per-lane async b128, W tile via TDM
// (2-D D#, data_size=2B, LDS padding 4 dwords / 16 dwords -> 40-half stride).
// MODE 0: QKV epilogue (RoPE, q-scale, V transposed).  MODE 1: fp32 store.
// ---------------------------------------------------------------------------
template <int MODE>
__global__ __launch_bounds__(256)
void gemm64x64_kernel(const _Float16* __restrict__ A, const _Float16* __restrict__ W,
                      _Float16* __restrict__ qout, _Float16* __restrict__ kout,
                      _Float16* __restrict__ vTout,
                      const float* __restrict__ rcos, const float* __restrict__ rsin,
                      float* __restrict__ outf) {
  __shared__ __align__(16) _Float16 As[2][64 * 40];
  __shared__ __align__(16) _Float16 Bs[2][64 * 40];
  __shared__ float Cs[64 * 65];

  const int tid = threadIdx.x;
  const int wv = tid >> 5;
  const int wm = wv >> 1;   // 0..3  (16-row band)
  const int wn = wv & 1;    // 0..1  (32-col band)
  const int mt = blockIdx.x * 64;
  const int jt = blockIdx.y * 64;
  const int ldrow = (tid & 127) >> 1;
  const int ldc = tid & 1;

#if HAVE_TDM
  auto tdm_issue = [&](int kk, int buf) {
    // Tensor DMA D#: 2-D tile 32(K) x 64(rows) of 2-byte elems from W,
    // pad +4 dwords every 16 dwords => LDS row stride 40 halves.
    const unsigned long long ga =
        (unsigned long long)(uintptr_t)(W + (size_t)jt * DIMX + kk);
    u32x4 g0;
    g0[0] = 1u;                                         // count=1, no gather
    g0[1] = (unsigned int)(uintptr_t)&Bs[buf][0];       // lds_addr (bytes)
    g0[2] = (unsigned int)ga;                           // global_addr[31:0]
    g0[3] = (unsigned int)((ga >> 32) & 0x1FFFFFFull) | (2u << 30);  // type=2
    i32x8 g1;
    g1[0] = (1 << 16) | (1 << 20) | (3 << 22) | (3 << 25);  // size=2B,pad en,16dw,+4dw
    g1[1] = (int)(1024u << 16);                         // tensor_dim0 = 1024
    g1[2] = (int)((unsigned)(MODE == 0 ? QKV_COLS : DIMX) << 16);  // tensor_dim1
    g1[3] = (int)(32u << 16);                           // tile_dim0 = 32
    g1[4] = 64;                                         // tile_dim1 = 64
    g1[5] = 1024;                                       // tensor_dim0_stride
    g1[6] = 0; g1[7] = 0;
    i32x4 g2 = {0, 0, 0, 0}, g3 = {0, 0, 0, 0};
    i32x8 g4 = {0, 0, 0, 0, 0, 0, 0, 0};                // unused 5th group (clang-23 arity)
    __builtin_amdgcn_tensor_load_to_lds(g0, g1, g2, g3, g4, 0);
  };
#endif

  auto stage = [&](int kk, int buf) {
    if (tid < 128)
      async_copy16(&As[buf][ldrow * 40 + ldc * 16],
                   A + (size_t)(mt + ldrow) * DIMX + kk + ldc * 16);
#if HAVE_TDM
    if (wv == 0) tdm_issue(kk, buf);
#else
    if (tid >= 128)
      async_copy16(&Bs[buf][ldrow * 40 + ldc * 16],
                   W + (size_t)(jt + ldrow) * DIMX + kk + ldc * 16);
#endif
  };

  v8f c00 = {}, c01 = {};
  stage(0, 0);
  for (int kk = 0, s = 0; kk < DIMX; kk += 32, ++s) {
    const int cur = s & 1;
    const bool more = (kk + 32) < DIMX;
    if (more) stage(kk + 32, cur ^ 1);   // overlap next-stage copy with math
    async_wait(more ? 1 : 0);
#if HAVE_TDM
    if (wv == 0) tensor_wait(more ? 1 : 0);
#endif
    __syncthreads();
    v16h af = load_frag_a(&As[cur][wm * 16 * 40], 40);
    v16h b0 = load_frag_b(&Bs[cur][(wn * 32 + 0) * 40], 40);
    v16h b1 = load_frag_b(&Bs[cur][(wn * 32 + 16) * 40], 40);
    c00 = wmma32(af, b0, c00);
    c01 = wmma32(af, b1, c01);
    __syncthreads();
  }

  if constexpr (MODE == 0) {
    store_c_lds(Cs + (wm * 16) * 65 + wn * 32, 65, c00);
    store_c_lds(Cs + (wm * 16) * 65 + wn * 32 + 16, 65, c01);
    __syncthreads();
    const int part = jt >> 10;            // 0=q 1=k 2=v
    const int head = (jt & 1023) >> 6;    // 64-col tile == one head
    for (int pi = tid; pi < 64 * 32; pi += 256) {
      const int lr = pi >> 5;
      const int dhe = (pi & 31) * 2;
      const float e = Cs[lr * 65 + dhe];
      const float o = Cs[lr * 65 + dhe + 1];
      const int m = mt + lr;
      const int b = m >> 11;
      const int n = m & 2047;
      const size_t bhh = (size_t)(b * HEADS + head);
      if (part == 2) {
        vTout[(bhh * DH + dhe) * N_SEQ + n] = (_Float16)e;
        vTout[(bhh * DH + dhe + 1) * N_SEQ + n] = (_Float16)o;
      } else {
        const float ce = rcos[n * DH + dhe], se = rsin[n * DH + dhe];
        const float co = rcos[n * DH + dhe + 1], so = rsin[n * DH + dhe + 1];
        float re = e * ce - o * se;       // x*cos + rotate_half(x)*sin
        float ro = o * co + e * so;
        if (part == 0) {
          re *= 0.125f; ro *= 0.125f;     // fold dh^-0.5 into q
          qout[(bhh * N_SEQ + n) * DH + dhe] = (_Float16)re;
          qout[(bhh * N_SEQ + n) * DH + dhe + 1] = (_Float16)ro;
        } else {
          kout[(bhh * N_SEQ + n) * DH + dhe] = (_Float16)re;
          kout[(bhh * N_SEQ + n) * DH + dhe + 1] = (_Float16)ro;
        }
      }
    }
  } else {
    const int lane = tid & 31;
    const int n = lane & 15;
    const int mb = (lane < 16) ? 0 : 8;
#pragma unroll
    for (int r = 0; r < 8; ++r) {
      const int m = mt + wm * 16 + mb + r;
      outf[(size_t)m * DIMX + jt + wn * 32 + n] = c00[r];
      outf[(size_t)m * DIMX + jt + wn * 32 + 16 + n] = c01[r];
    }
  }
}

// ---------------------------------------------------------------------------
// K3: flash attention. 4 waves/block share one (b,h); K/V tiles (32 keys)
// double-buffered in LDS via async copies, reused by all 4 waves.
// S = Q Kt via WMMA -> online softmax (DPP row reductions) -> P via LDS
// bounce into A-layout -> O += P V. gate & 1/l in epilogue.
// ---------------------------------------------------------------------------
__global__ __launch_bounds__(128)
void flash_kernel(const _Float16* __restrict__ q, const _Float16* __restrict__ k,
                  const _Float16* __restrict__ vT, const float* __restrict__ gates,
                  _Float16* __restrict__ attn) {
  __shared__ __align__(16) _Float16 Ks[2][32 * 72];   // [key][dh], 72-half stride
  __shared__ __align__(16) _Float16 Vs[2][64 * 40];   // [dh][key], 40-half stride
  __shared__ __align__(16) _Float16 plds[4][16 * 40];

  const int tid = threadIdx.x;
  const int wv = tid >> 5;
  const int lane = tid & 31;
  const int bh = blockIdx.x >> 5;             // 32 blocks per (b,h)
  const int qt = (blockIdx.x & 31) * 4 + wv;  // 16-row q tile per wave
  const int b = bh >> 4, h = bh & 15;

  const _Float16* qb = q + ((size_t)bh * N_SEQ + qt * 16) * DH;
  const _Float16* kb = k + (size_t)bh * N_SEQ * DH;
  const _Float16* vb = vT + (size_t)bh * DH * N_SEQ;

  const int kr = tid >> 2, kc = tid & 3;   // Ks: 32 rows x 4 chunks
  const int vr = tid >> 1, vc = tid & 1;   // Vs: 64 rows x 2 chunks

  const v16h qa0 = load_frag_a(qb, DH);       // dh 0..31
  const v16h qa1 = load_frag_a(qb + 32, DH);  // dh 32..63

  v8f o0 = {}, o1 = {}, o2 = {}, o3 = {};
  float mrow[8], lrow[8];
#pragma unroll
  for (int r = 0; r < 8; ++r) { mrow[r] = -3.0e38f; lrow[r] = 0.f; }
  const int mb = (lane < 16) ? 0 : 8;
  const int ncol = lane & 15;
  _Float16* ps = &plds[wv][0];

  auto stage = [&](int j, int buf) {
    async_copy16(&Ks[buf][kr * 72 + kc * 16], kb + (size_t)(j + kr) * DH + kc * 16);
    async_copy16(&Vs[buf][vr * 40 + vc * 16], vb + (size_t)vr * N_SEQ + j + vc * 16);
  };

  stage(0, 0);
  for (int j = 0, s = 0; j < N_SEQ; j += 32, ++s) {
    const int cur = s & 1;
    const bool more = (j + 32) < N_SEQ;
    if (more) stage(j + 32, cur ^ 1);
    async_wait(more ? 2 : 0);
    __syncthreads();
    const _Float16* Kc = &Ks[cur][0];
    const _Float16* Vc = &Vs[cur][0];

    v16h kb00 = load_frag_b(Kc, 72);             // keys j..j+15,  dh 0..31
    v16h kb01 = load_frag_b(Kc + 32, 72);        // keys j..j+15,  dh 32..63
    v16h kb10 = load_frag_b(Kc + 16 * 72, 72);   // keys j+16..31, dh 0..31
    v16h kb11 = load_frag_b(Kc + 16 * 72 + 32, 72);
    v8f s0 = {}, s1 = {};
    s0 = wmma32(qa0, kb00, s0); s0 = wmma32(qa1, kb01, s0);
    s1 = wmma32(qa0, kb10, s1); s1 = wmma32(qa1, kb11, s1);

    float p0[8], p1[8];
#pragma unroll
    for (int r = 0; r < 8; ++r) {
      const float mx = red16_max(fmaxf(s0[r], s1[r]));
      const float mn = fmaxf(mrow[r], mx);
      const float alpha = __expf(mrow[r] - mn);
      p0[r] = __expf(s0[r] - mn);
      p1[r] = __expf(s1[r] - mn);
      lrow[r] = lrow[r] * alpha + red16_sum(p0[r] + p1[r]);
      mrow[r] = mn;
      o0[r] *= alpha; o1[r] *= alpha; o2[r] *= alpha; o3[r] *= alpha;
    }

    // P (C-layout) -> LDS -> A-fragment layout (wave-private scratch)
#pragma unroll
    for (int r = 0; r < 8; ++r) {
      ps[(mb + r) * 40 + ncol] = (_Float16)p0[r];
      ps[(mb + r) * 40 + 16 + ncol] = (_Float16)p1[r];
    }
    asm volatile("s_wait_dscnt 0" ::: "memory");
    const v16h pa = load_frag_a(ps, 40);

    o0 = wmma32(pa, load_frag_b(Vc + 0 * 16 * 40, 40), o0);
    o1 = wmma32(pa, load_frag_b(Vc + 1 * 16 * 40, 40), o1);
    o2 = wmma32(pa, load_frag_b(Vc + 2 * 16 * 40, 40), o2);
    o3 = wmma32(pa, load_frag_b(Vc + 3 * 16 * 40, 40), o3);
    __syncthreads();   // all waves done with this buffer before re-staging
  }

  // epilogue: 1/l, sigmoid gate, store f16 row-major (b*n, h*64+dh)
#pragma unroll
  for (int r = 0; r < 8; ++r) {
    const int nrow = qt * 16 + mb + r;
    const float g = gates[((size_t)b * N_SEQ + nrow) * HEADS + h];
    const float sc = g / lrow[r];
    const size_t base = ((size_t)b * N_SEQ + nrow) * DIMX + h * DH;
    attn[base + 0 * 16 + ncol] = (_Float16)(o0[r] * sc);
    attn[base + 1 * 16 + ncol] = (_Float16)(o1[r] * sc);
    attn[base + 2 * 16 + ncol] = (_Float16)(o2[r] * sc);
    attn[base + 3 * 16 + ncol] = (_Float16)(o3[r] * sc);
  }
}

// ---------------------------------------------------------------------------
// launch
// ---------------------------------------------------------------------------
extern "C" void kernel_launch(void* const* d_in, const int* in_sizes, int n_in,
                              void* d_out, int out_size, void* d_ws, size_t ws_size,
                              hipStream_t stream) {
  const float* x     = (const float*)d_in[0];
  const float* rcos  = (const float*)d_in[1];
  const float* rsin  = (const float*)d_in[2];
  const float* gamma = (const float*)d_in[3];
  const float* wqkv  = (const float*)d_in[4];
  const float* wg    = (const float*)d_in[5];
  const float* bg    = (const float*)d_in[6];
  const float* wout  = (const float*)d_in[7];
  float* out = (float*)d_out;

  char* ws = (char*)d_ws;
  _Float16* xnh   = (_Float16*)ws;                 ws += (size_t)ROWS * DIMX * 2;
  _Float16* wqkvh = (_Float16*)ws;                 ws += (size_t)QKV_COLS * DIMX * 2;
  _Float16* wouth = (_Float16*)ws;                 ws += (size_t)DIMX * DIMX * 2;
  _Float16* qh    = (_Float16*)ws;                 ws += (size_t)B_SZ * HEADS * N_SEQ * DH * 2;
  _Float16* kh    = (_Float16*)ws;                 ws += (size_t)B_SZ * HEADS * N_SEQ * DH * 2;
  _Float16* vTh   = (_Float16*)ws;                 ws += (size_t)B_SZ * HEADS * DH * N_SEQ * 2;
  float*    gatesf= (float*)ws;                    ws += (size_t)ROWS * HEADS * 4;
  _Float16* attnh = (_Float16*)ws;                 ws += (size_t)ROWS * DIMX * 2;

  cvt_f16_kernel<<<(QKV_COLS * DIMX + 255) / 256, 256, 0, stream>>>(wqkv, wqkvh, QKV_COLS * DIMX);
  cvt_f16_kernel<<<(DIMX * DIMX + 255) / 256, 256, 0, stream>>>(wout, wouth, DIMX * DIMX);

  rmsnorm_gates_kernel<<<ROWS, 256, 0, stream>>>(x, gamma, wg, bg, xnh, gatesf);

  dim3 gq(ROWS / 64, QKV_COLS / 64);
  gemm64x64_kernel<0><<<gq, 256, 0, stream>>>(xnh, wqkvh, qh, kh, vTh, rcos, rsin, nullptr);

  flash_kernel<<<(B_SZ * HEADS * (N_SEQ / 16)) / 4, 128, 0, stream>>>(qh, kh, vTh, gatesf, attnh);

  dim3 go(ROWS / 64, DIMX / 64);
  gemm64x64_kernel<1><<<go, 256, 0, stream>>>(attnh, wouth, nullptr, nullptr, nullptr,
                                              nullptr, nullptr, out);
}
